// EncoderBlock_75179107549385
// MI455X (gfx1250) — matrix operand order repaired and checked by reading, not compiled, not run on hardware
//
#include <hip/hip_runtime.h>
#include <hip/hip_bf16.h>

typedef __attribute__((ext_vector_type(16))) _Float16 v16h;
typedef __attribute__((ext_vector_type(8)))  _Float16 v8h;
typedef __attribute__((ext_vector_type(8)))  float    v8f;

#define DEV __device__ __forceinline__

static constexpr int SEQ = 2048;
static constexpr int DM  = 512;
static constexpr int NH  = 8;
static constexpr int DK  = 64;
static constexpr int DFF = 2048;
static constexpr int MROWS = 4 * SEQ;   // B*S = 8192

// ---------------------------------------------------------------------------
// WMMA helper (CDNA5 wave32, V_WMMA_F32_16X16X32_F16)
// ---------------------------------------------------------------------------
DEV v8f wmma16(v16h a, v16h b, v8f c) {
  return __builtin_amdgcn_wmma_f32_16x16x32_f16(false, a, false, b, (short)0, c,
                                                false, false);
}

// A fragment (M=16 x K=32, f16) from row-major source, leading dim ld (halves).
// Layout: row=lane&15; contraction = (lane>>4)*16+i.
// One contiguous 16-half run -> 2x ds_load_b128 when 16B aligned.
DEV v16h load_a(const _Float16* base, int ld) {
  const int lane = threadIdx.x & 31;
  const _Float16* p = base + (size_t)(lane & 15) * ld + ((lane >> 4) * 16);
  v16h a;
#pragma unroll
  for (int i = 0; i < 16; ++i) a[i] = p[i];
  return a;
}

// B fragment (K=32 x N=16) from N-major (transposed) source Bt[n][k], ld in
// halves. Layout used here (same k order as load_a): lane = column N (lane&15); contraction = (lane>>4)*16+i.
// One contiguous 16-half run -> 2x ds_load_b128.
DEV v16h load_bt(const _Float16* base, int ld) {
  const int lane = threadIdx.x & 31;
  const _Float16* p = base + (size_t)(lane & 15) * ld + ((lane >> 4) * 16);
  v16h b;
#pragma unroll
  for (int i = 0; i < 16; ++i) b[i] = p[i];
  return b;
}

// Half-wave (16-lane) reductions matching the C-fragment row striping.
DEV float redmax16(float v) {
#pragma unroll
  for (int o = 1; o < 16; o <<= 1) v = fmaxf(v, __shfl_xor(v, o, 32));
  return v;
}
DEV float redsum16(float v) {
#pragma unroll
  for (int o = 1; o < 16; o <<= 1) v += __shfl_xor(v, o, 32);
  return v;
}

// gfx1250 async global->LDS copy (16B per lane), tracked by ASYNCcnt.
DEV void async_copy_b128(const _Float16* gsrc, _Float16* lds_dst) {
  const unsigned lds_off = (unsigned)(size_t)lds_dst;  // low 32b = LDS address
  const unsigned long long ga = (unsigned long long)(size_t)gsrc;
  asm volatile("global_load_async_to_lds_b128 %0, %1, off"
               :: "v"(lds_off), "v"(ga) : "memory");
}
DEV void async_wait() {
  asm volatile("s_wait_asynccnt 0x0" ::: "memory");
}

// ---------------------------------------------------------------------------
// fp32 -> f16 conversion (weights)
// ---------------------------------------------------------------------------
__global__ void cvt_f32_f16(const float* __restrict__ in,
                            _Float16* __restrict__ out, int n) {
  int i = blockIdx.x * blockDim.x + threadIdx.x;
  if (i < n) out[i] = (_Float16)in[i];
}

// ---------------------------------------------------------------------------
// LayerNorm: one wave per 512-wide row; unbiased std (ddof=1), /(std+eps)
// ---------------------------------------------------------------------------
__global__ __launch_bounds__(256) void layernorm_k(
    const float* __restrict__ x, const float* __restrict__ g,
    const float* __restrict__ be, _Float16* __restrict__ out, int rows) {
  const int row = blockIdx.x * 8 + (threadIdx.x >> 5);
  if (row >= rows) return;
  const int lane = threadIdx.x & 31;
  const float* xr = x + (size_t)row * DM;
  float v[16];
  float s = 0.f;
#pragma unroll
  for (int i = 0; i < 16; ++i) { v[i] = xr[lane + 32 * i]; s += v[i]; }
#pragma unroll
  for (int o = 16; o >= 1; o >>= 1) s += __shfl_xor(s, o, 32);
  const float mean = s * (1.0f / DM);
  float q = 0.f;
#pragma unroll
  for (int i = 0; i < 16; ++i) { float d = v[i] - mean; q += d * d; }
#pragma unroll
  for (int o = 16; o >= 1; o >>= 1) q += __shfl_xor(q, o, 32);
  const float rstd = 1.0f / (sqrtf(q * (1.0f / (DM - 1))) + 1e-6f);
  _Float16* orow = out + (size_t)row * DM;
#pragma unroll
  for (int i = 0; i < 16; ++i) {
    const int c = lane + 32 * i;
    orow[c] = (_Float16)(g[c] * (v[i] - mean) * rstd + be[c]);
  }
}

// ---------------------------------------------------------------------------
// Tiled WMMA GEMM: C[M,N] = A[M,K] @ B[K,N] (f16 in, f32 acc) + bias.
// 128 threads = 4 waves; tile 128x64; BK=32; wave owns 32x64 (8 WMMAs/step).
// Double-buffered LDS: A via async global->LDS issued one step ahead; B loaded
// to VGPRs ahead, scatter-stored transposed (N-major) after the WMMAs.
// OUTMODE: 0 = fp32 row-major, 1 = f16 row-major, 2 = f16 scatter [B,H,S,DK]
// ---------------------------------------------------------------------------
template <int OUTMODE, bool RELU, bool RESID>
__global__ __launch_bounds__(128) void gemm_wmma(
    const _Float16* __restrict__ A, const _Float16* __restrict__ B,
    const float* __restrict__ bias, const float* __restrict__ resid,
    float* __restrict__ outF, _Float16* __restrict__ outH,
    int M, int N, int K) {
  __shared__ _Float16 As[2][128][40];  // row-major M x K slice (+8 pad)
  __shared__ _Float16 Bt[2][64][40];   // transposed: [n][k]

  const int tid  = threadIdx.x;
  const int wv   = tid >> 5;
  const int lane = tid & 31;
  const int m0   = blockIdx.y * 128;
  const int n0   = blockIdx.x * 64;

  auto issueA = [&](int buf, int k0) {
#pragma unroll
    for (int l = 0; l < 4; ++l) {
      const int ch = tid + l * 128;             // 512 chunks of 8 halves
      const int r = ch >> 2, c8 = (ch & 3) * 8;
      async_copy_b128(&A[(size_t)(m0 + r) * K + k0 + c8], &As[buf][r][c8]);
    }
  };
  auto loadB = [&](int k0, v8h* bv) {
#pragma unroll
    for (int l = 0; l < 2; ++l) {
      const int idx = tid + l * 128;            // 256 chunks of 8 halves
      const int r = idx >> 3, c8 = (idx & 7) * 8;
      bv[l] = *(const v8h*)&B[(size_t)(k0 + r) * N + n0 + c8];
    }
  };
  auto storeBt = [&](int buf, const v8h* bv) {
#pragma unroll
    for (int l = 0; l < 2; ++l) {
      const int idx = tid + l * 128;
      const int r = idx >> 3, c8 = (idx & 7) * 8;
#pragma unroll
      for (int i = 0; i < 8; ++i) Bt[buf][c8 + i][r] = bv[l][i];
    }
  };

  v8f z = {};
  v8f acc[2][4];
#pragma unroll
  for (int s = 0; s < 2; ++s)
#pragma unroll
    for (int f = 0; f < 4; ++f) acc[s][f] = z;

  // prologue: fill buffer 0
  issueA(0, 0);
  {
    v8h bv[2];
    loadB(0, bv);
    storeBt(0, bv);
  }

  int cur = 0;
  for (int k0 = 0; k0 < K; k0 += 32) {
    async_wait();        // tile `cur` A-copy landed (this wave's)
    __syncthreads();     // all waves: tile `cur` complete, prev reads done

    const bool nxt = (k0 + 32 < K);
    v8h bv[2];
    if (nxt) {           // start next tile while we compute
      issueA(1 - cur, k0 + 32);
      loadB(k0 + 32, bv);
    }

    const v16h a0 = load_a(&As[cur][wv * 32][0], 40);
    const v16h a1 = load_a(&As[cur][wv * 32 + 16][0], 40);
#pragma unroll
    for (int f = 0; f < 4; ++f) {
      const v16h b = load_bt(&Bt[cur][f * 16][0], 40);
      acc[0][f] = wmma16(a0, b, acc[0][f]);
      acc[1][f] = wmma16(a1, b, acc[1][f]);
    }

    if (nxt) storeBt(1 - cur, bv);  // scatter after compute, into idle buffer
    cur ^= 1;
  }

  // epilogue (C-frag: lane = col (lane&15), VGPR r = row r + 8*(lane>>4))
  const int col = lane & 15, hi = lane >> 4;
#pragma unroll
  for (int s = 0; s < 2; ++s) {
#pragma unroll
    for (int f = 0; f < 4; ++f) {
      const int n = n0 + f * 16 + col;
      const float bvx = bias[n];
#pragma unroll
      for (int r = 0; r < 8; ++r) {
        const int gm = m0 + wv * 32 + s * 16 + r + 8 * hi;
        float v = acc[s][f][r] + bvx;
        if (RELU) v = fmaxf(v, 0.0f);
        if (RESID) v += resid[(size_t)gm * N + n];
        if (OUTMODE == 0) {
          outF[(size_t)gm * N + n] = v;
        } else if (OUTMODE == 1) {
          outH[(size_t)gm * N + n] = (_Float16)v;
        } else {  // QKV scatter -> [B, H, S, DK]
          const int b = gm >> 11, q = gm & (SEQ - 1);
          const int h = n >> 6, d = n & (DK - 1);
          outH[(((size_t)b * NH + h) * SEQ + q) * DK + d] = (_Float16)v;
        }
      }
    }
  }
}

// ---------------------------------------------------------------------------
// Flash attention: grid (S/64, B*H), 128 threads (4 waves, 16 q-rows each).
// scores = floor(QK^T / 8); mask==0 -> -1e9; online softmax; O = P @ V.
// Double-buffered tiles: K async global->LDS one step ahead; V loaded to VGPRs
// ahead and scatter-stored transposed after the WMMAs.
// ---------------------------------------------------------------------------
__global__ __launch_bounds__(128) void attention_k(
    const _Float16* __restrict__ Qh, const _Float16* __restrict__ Kh,
    const _Float16* __restrict__ Vh, const int* __restrict__ mask,
    _Float16* __restrict__ ctx) {
  __shared__ _Float16 Ks[2][32][72];   // row-major [key][d]
  __shared__ _Float16 Vt[2][64][40];   // transposed [d][key]
  __shared__ _Float16 Pl[4][16][40];   // per-wave P staging

  const int tid = threadIdx.x, wv = tid >> 5, lane = tid & 31;
  const int col = lane & 15, hi = lane >> 4;
  const int bh = blockIdx.y;  // b*8 + h
  const int q0 = blockIdx.x * 64 + wv * 16;
  const size_t base = (size_t)bh * SEQ * DK;
  const int b = bh >> 3, h = bh & 7;

  auto issueK = [&](int buf, int j) {
#pragma unroll
    for (int t = 0; t < 2; ++t) {
      const int idx = tid + t * 128;  // 256 chunks of 8 halves
      const int kl = idx >> 3, c8 = (idx & 7) * 8;
      async_copy_b128(&Kh[base + (size_t)(j + kl) * DK + c8], &Ks[buf][kl][c8]);
    }
  };
  auto loadV = [&](int j, v8h* vv) {
#pragma unroll
    for (int t = 0; t < 2; ++t) {
      const int idx = tid + t * 128;
      const int kl = idx >> 3, c8 = (idx & 7) * 8;
      vv[t] = *(const v8h*)&Vh[base + (size_t)(j + kl) * DK + c8];
    }
  };
  auto storeVt = [&](int buf, const v8h* vv) {
#pragma unroll
    for (int t = 0; t < 2; ++t) {
      const int idx = tid + t * 128;
      const int kl = idx >> 3, c8 = (idx & 7) * 8;
#pragma unroll
      for (int i = 0; i < 8; ++i) Vt[buf][c8 + i][kl] = vv[t][i];
    }
  };

  // Q strip (16 x 64) as two A-fragments kept in registers
  v16h qa[2];
#pragma unroll
  for (int f = 0; f < 2; ++f)
    qa[f] = load_a(Qh + base + (size_t)q0 * DK + f * 32, DK);

  v8f z = {};
  v8f o[4];
#pragma unroll
  for (int f = 0; f < 4; ++f) o[f] = z;
  float m[8], l[8];
#pragma unroll
  for (int r = 0; r < 8; ++r) { m[r] = -3.0e38f; l[r] = 0.f; }

  // prologue: fill buffer 0
  issueK(0, 0);
  {
    v8h vv[2];
    loadV(0, vv);
    storeVt(0, vv);
  }

  int cur = 0;
  for (int j = 0; j < SEQ; j += 32) {
    async_wait();
    __syncthreads();

    const bool nxt = (j + 32 < SEQ);
    v8h vv[2];
    if (nxt) {
      issueK(1 - cur, j + 32);
      loadV(j + 32, vv);
    }

    // two 16-key score tiles: S = Q (16x64) @ K^T (64x16)
    float sc[2][8];
#pragma unroll
    for (int t = 0; t < 2; ++t) {
      v8f s = z;
#pragma unroll
      for (int kk = 0; kk < 2; ++kk) {
        // B-frag from row-major K tile: column = key, contraction = d
        const _Float16* kp = &Ks[cur][t * 16 + col][kk * 32 + hi * 16];
        v16h bf;
#pragma unroll
        for (int i = 0; i < 16; ++i) bf[i] = kp[i];
        s = wmma16(qa[kk], bf, s);
      }
#pragma unroll
      for (int r = 0; r < 8; ++r) {
        float v = floorf(s[r] * 0.125f);  // floor_divide(qk, sqrt(64))
        const int key = j + t * 16 + col;
        if (mask[(size_t)(q0 + r + 8 * hi) * SEQ + key] == 0) v = -1.0e9f;
        sc[t][r] = v;
      }
    }

    // online softmax update over these 32 keys
#pragma unroll
    for (int r = 0; r < 8; ++r) {
      const float tm = redmax16(fmaxf(sc[0][r], sc[1][r]));
      const float nm = fmaxf(m[r], tm);
      const float sca = __expf(m[r] - nm);
      m[r] = nm;
      const float p0 = __expf(sc[0][r] - nm);
      const float p1 = __expf(sc[1][r] - nm);
      sc[0][r] = p0; sc[1][r] = p1;
      l[r] = l[r] * sca + redsum16(p0 + p1);
#pragma unroll
      for (int f = 0; f < 4; ++f) o[f][r] *= sca;
    }

    // stage P (16x32) into per-wave LDS, reload as contiguous A-fragment
#pragma unroll
    for (int t = 0; t < 2; ++t)
#pragma unroll
      for (int r = 0; r < 8; ++r)
        Pl[wv][r + 8 * hi][t * 16 + col] = (_Float16)sc[t][r];

    const v16h pa = load_a(&Pl[wv][0][0], 40);
#pragma unroll
    for (int f = 0; f < 4; ++f) {
      // V B-frag from Vt[d][key]: contiguous -> 2x ds_load_b128
      const v16h bf = load_bt(&Vt[cur][f * 16][0], 40);
      o[f] = wmma16(pa, bf, o[f]);
    }

    if (nxt) storeVt(1 - cur, vv);  // scatter after compute, into idle buffer
    cur ^= 1;
  }

  // normalize and store ctx in (B, S, D_MODEL) layout (head-concat)
#pragma unroll
  for (int r = 0; r < 8; ++r) {
    const float inv = 1.0f / l[r];
    const int qg = q0 + r + 8 * hi;
    const size_t off = ((size_t)b * SEQ + qg) * DM + h * DK;
#pragma unroll
    for (int f = 0; f < 4; ++f)
      ctx[off + f * 16 + col] = (_Float16)(o[f][r] * inv);
  }
}

// ---------------------------------------------------------------------------
// Host-side orchestration
// ---------------------------------------------------------------------------
extern "C" void kernel_launch(void* const* d_in, const int* in_sizes, int n_in,
                              void* d_out, int out_size, void* d_ws, size_t ws_size,
                              hipStream_t stream) {
  const float* x   = (const float*)d_in[0];
  const int*   msk = (const int*)d_in[1];
  const float* wq  = (const float*)d_in[2];
  const float* bq  = (const float*)d_in[3];
  const float* wk  = (const float*)d_in[4];
  const float* bk  = (const float*)d_in[5];
  const float* wv  = (const float*)d_in[6];
  const float* bv  = (const float*)d_in[7];
  const float* wo  = (const float*)d_in[8];
  const float* bo  = (const float*)d_in[9];
  const float* w1  = (const float*)d_in[10];
  const float* b1  = (const float*)d_in[11];
  const float* w2  = (const float*)d_in[12];
  const float* b2  = (const float*)d_in[13];
  const float* g1  = (const float*)d_in[14];
  const float* be1 = (const float*)d_in[15];
  const float* g2  = (const float*)d_in[16];
  const float* be2 = (const float*)d_in[17];
  float* out = (float*)d_out;

  char* ws = (char*)d_ws;
  const size_t KB = 1024, MB = 1024 * 1024;
  _Float16* wqh = (_Float16*)(ws + 0);
  _Float16* wkh = (_Float16*)(ws + 512 * KB);
  _Float16* wvh = (_Float16*)(ws + 1 * MB);
  _Float16* woh = (_Float16*)(ws + 1 * MB + 512 * KB);
  _Float16* w1h = (_Float16*)(ws + 2 * MB);
  _Float16* w2h = (_Float16*)(ws + 4 * MB);
  _Float16* xnh = (_Float16*)(ws + 6 * MB);            // 8 MB (LN1 then LN2)
  _Float16* Qh  = (_Float16*)(ws + 14 * MB);           // 8 MB
  _Float16* Kh  = (_Float16*)(ws + 22 * MB);           // 8 MB
  _Float16* Vh  = (_Float16*)(ws + 30 * MB);           // 8 MB
  _Float16* ctxh= (_Float16*)(ws + 38 * MB);           // 8 MB
  float*    x2  = (float*)   (ws + 46 * MB);           // 16 MB fp32
  _Float16* Hh  = (_Float16*)(ws + 62 * MB);           // 32 MB

  // 1) weights -> f16
  cvt_f32_f16<<<(DM * DM + 255) / 256, 256, 0, stream>>>(wq, wqh, DM * DM);
  cvt_f32_f16<<<(DM * DM + 255) / 256, 256, 0, stream>>>(wk, wkh, DM * DM);
  cvt_f32_f16<<<(DM * DM + 255) / 256, 256, 0, stream>>>(wv, wvh, DM * DM);
  cvt_f32_f16<<<(DM * DM + 255) / 256, 256, 0, stream>>>(wo, woh, DM * DM);
  cvt_f32_f16<<<(DM * DFF + 255) / 256, 256, 0, stream>>>(w1, w1h, DM * DFF);
  cvt_f32_f16<<<(DFF * DM + 255) / 256, 256, 0, stream>>>(w2, w2h, DFF * DM);

  // 2) LN1
  layernorm_k<<<MROWS / 8, 256, 0, stream>>>(x, g1, be1, xnh, MROWS);

  // 3) Q, K, V projections (scatter to [B,H,S,64] f16)
  dim3 g512(DM / 64, MROWS / 128);
  gemm_wmma<2, false, false><<<g512, 128, 0, stream>>>(xnh, wqh, bq, nullptr,
      nullptr, Qh, MROWS, DM, DM);
  gemm_wmma<2, false, false><<<g512, 128, 0, stream>>>(xnh, wkh, bk, nullptr,
      nullptr, Kh, MROWS, DM, DM);
  gemm_wmma<2, false, false><<<g512, 128, 0, stream>>>(xnh, wvh, bv, nullptr,
      nullptr, Vh, MROWS, DM, DM);

  // 4) flash attention -> ctx (B,S,512) f16
  attention_k<<<dim3(SEQ / 64, 4 * NH), 128, 0, stream>>>(Qh, Kh, Vh, msk, ctxh);

  // 5) output projection + residual: x2 = x + ctx @ wo + bo (fp32)
  gemm_wmma<0, false, true><<<g512, 128, 0, stream>>>(ctxh, woh, bo, x,
      x2, nullptr, MROWS, DM, DM);

  // 6) LN2
  layernorm_k<<<MROWS / 8, 256, 0, stream>>>(x2, g2, be2, xnh, MROWS);

  // 7) FF1: H = relu(xn2 @ w1 + b1) (f16)
  gemm_wmma<1, true, false><<<dim3(DFF / 64, MROWS / 128), 128, 0, stream>>>(
      xnh, w1h, b1, nullptr, nullptr, Hh, MROWS, DFF, DM);

  // 8) FF2: out = x2 + H @ w2 + b2 (fp32)
  gemm_wmma<0, false, true><<<g512, 128, 0, stream>>>(Hh, w2h, b2, x2,
      out, nullptr, MROWS, DM, DFF);
}
